// AttentionBlock_30666066493470
// MI455X (gfx1250) — matrix operand order repaired and checked by reading, not compiled, not run on hardware
//
#include <hip/hip_runtime.h>
#include <math.h>

typedef __bf16 bf16t;
typedef __attribute__((ext_vector_type(16))) __bf16 v16bf;
typedef __attribute__((ext_vector_type(8)))  __bf16 v8bf;
typedef __attribute__((ext_vector_type(8)))  float  v8f;

namespace {
constexpr int kBatch  = 8;
constexpr int kC      = 512;    // channels
constexpr int kN      = 1024;   // h*w tokens
constexpr int kGroups = 32;
constexpr float kScale = 0.04419417382415922f;  // 512^-0.5
constexpr float kEps   = 1e-5f;
}

// ---------- helpers ----------
__device__ inline bf16t f2bf(float f) {
  union { float f; unsigned u; } a; a.f = f;
  unsigned r = a.u + 0x7FFFu + ((a.u >> 16) & 1u);   // round-to-nearest-even
  union { unsigned short s; bf16t b; } o; o.s = (unsigned short)(r >> 16);
  return o.b;
}

// A fragment: 16x32 bf16, A row-major [M,K].
// lane<16: row=lane,   K in {0..7} U {16..23}
// lane>=16: row=lane-16, K in {8..15} U {24..31}
__device__ inline v16bf load_a_frag(const bf16t* __restrict__ A, int lda,
                                    int row, int k0, int lane) {
  const bf16t* p = A + (size_t)(row + (lane & 15)) * lda + k0 + ((lane >> 4) << 3);
  union { v16bf v; v8bf h[2]; } u;
  u.h[0] = *(const v8bf*)(p);
  u.h[1] = *(const v8bf*)(p + 16);
  return u.v;
}

// B fragment: 32x16 bf16, B supplied transposed: Bt row-major [N,K].
// lane<16: col=lane, K in {0..7} U {16..23} ; lane>=16: col=lane-16, K in {8..15} U {24..31}
__device__ inline v16bf load_b_frag(const bf16t* __restrict__ Bt, int ldb,
                                    int col, int k0, int lane) {
  const bf16t* p = Bt + (size_t)(col + (lane & 15)) * ldb + k0 + ((lane >> 4) << 3);
  union { v16bf v; v8bf h[2]; } u;
  u.h[0] = *(const v8bf*)(p);
  u.h[1] = *(const v8bf*)(p + 16);
  return u.v;
}

// One wave computes a 32x64 output tile (2x4 of 16x16 WMMA tiles), f32 accum.
// 12 b128 loads per 8 WMMAs -> 1.5 loads/WMMA.
__device__ inline void wave_tile_mma(const bf16t* __restrict__ A, int lda,
                                     const bf16t* __restrict__ Bt, int ldb,
                                     int K, int row0, int col0, int lane,
                                     v8f acc[2][4]) {
  for (int k0 = 0; k0 < K; k0 += 32) {
    v16bf a0 = load_a_frag(A, lda, row0,      k0, lane);
    v16bf a1 = load_a_frag(A, lda, row0 + 16, k0, lane);
    v16bf b0 = load_b_frag(Bt, ldb, col0,      k0, lane);
    v16bf b1 = load_b_frag(Bt, ldb, col0 + 16, k0, lane);
    v16bf b2 = load_b_frag(Bt, ldb, col0 + 32, k0, lane);
    v16bf b3 = load_b_frag(Bt, ldb, col0 + 48, k0, lane);
    if (k0 + 32 < K) {  // prefetch next K slab -> global_prefetch_b8
      __builtin_prefetch(A  + (size_t)(row0 + (lane & 15)) * lda + k0 + 32, 0, 1);
      __builtin_prefetch(Bt + (size_t)(col0 + (lane & 15)) * ldb + k0 + 32, 0, 1);
      __builtin_prefetch(Bt + (size_t)(col0 + 32 + (lane & 15)) * ldb + k0 + 32, 0, 1);
    }
    acc[0][0] = __builtin_amdgcn_wmma_f32_16x16x32_bf16(false, a0, false, b0, (short)0, acc[0][0], false, false);
    acc[0][1] = __builtin_amdgcn_wmma_f32_16x16x32_bf16(false, a0, false, b1, (short)0, acc[0][1], false, false);
    acc[0][2] = __builtin_amdgcn_wmma_f32_16x16x32_bf16(false, a0, false, b2, (short)0, acc[0][2], false, false);
    acc[0][3] = __builtin_amdgcn_wmma_f32_16x16x32_bf16(false, a0, false, b3, (short)0, acc[0][3], false, false);
    acc[1][0] = __builtin_amdgcn_wmma_f32_16x16x32_bf16(false, a1, false, b0, (short)0, acc[1][0], false, false);
    acc[1][1] = __builtin_amdgcn_wmma_f32_16x16x32_bf16(false, a1, false, b1, (short)0, acc[1][1], false, false);
    acc[1][2] = __builtin_amdgcn_wmma_f32_16x16x32_bf16(false, a1, false, b2, (short)0, acc[1][2], false, false);
    acc[1][3] = __builtin_amdgcn_wmma_f32_16x16x32_bf16(false, a1, false, b3, (short)0, acc[1][3], false, false);
  }
}

__device__ inline void zero_acc(v8f acc[2][4]) {
  v8f z = {0.f, 0.f, 0.f, 0.f, 0.f, 0.f, 0.f, 0.f};
  for (int i = 0; i < 2; ++i)
    for (int j = 0; j < 4; ++j) acc[i][j] = z;
}

// C/D 16x16 layout: VGPR e, lanes 0-15 -> M=e, N=lane; lanes 16-31 -> M=e+8, N=lane-16
#define ACC_ROW(si, e) ((si) * 16 + (((lane) >> 4) << 3) + (e))
#define ACC_COL(sj)    ((sj) * 16 + ((lane) & 15))

// wave mapping inside a 256-thread block: 4 waves along M x 2 waves along N
// block tile = 128 (M) x 128 (N); wave tile = 32 x 64
#define WAVE_COORDS()                                            \
  const int lane = threadIdx.x & 31, wave = threadIdx.x >> 5;    \
  const int row0 = blockIdx.x * 128 + (wave & 3) * 32;           \
  const int col0 = blockIdx.y * 128 + (wave >> 2) * 64;

// ---------- GroupNorm: x[b,c,h,w] f32 -> xnT[b][n][c] bf16 ----------
__global__ __launch_bounds__(256) void gn_kernel(const float* __restrict__ x,
                                                 const float* __restrict__ w,
                                                 const float* __restrict__ bias,
                                                 bf16t* __restrict__ xnT) {
  const int b = blockIdx.x >> 5;     // 8 batches
  const int g = blockIdx.x & 31;     // 32 groups (16 ch each)
  const int tid = threadIdx.x;
  const float* xb = x + ((size_t)b * kC + (size_t)g * 16) * kN;
  float s = 0.f, ss = 0.f;
  for (int i = tid; i < 16 * kN; i += 256) { float v = xb[i]; s += v; ss += v * v; }
  __shared__ float shs[256], shq[256];
  shs[tid] = s; shq[tid] = ss;
  __syncthreads();
  for (int off = 128; off > 0; off >>= 1) {
    if (tid < off) { shs[tid] += shs[tid + off]; shq[tid] += shq[tid + off]; }
    __syncthreads();
  }
  const float inv_n = 1.0f / (16.f * (float)kN);
  float mean = shs[0] * inv_n;
  float var  = shq[0] * inv_n - mean * mean;
  float rstd = rsqrtf(var + kEps);
  bf16t* xo = xnT + (size_t)b * kN * kC;
  for (int i = tid; i < 16 * kN; i += 256) {
    int c = g * 16 + (i >> 10);
    int n = i & (kN - 1);
    float v = (xb[i] - mean) * rstd * w[c] + bias[c];
    xo[(size_t)n * kC + c] = f2bf(v);
  }
}

// ---------- weight f32 -> bf16 ----------
__global__ void wconv_kernel(const float* __restrict__ a, bf16t* __restrict__ oa, int na,
                             const float* __restrict__ bsrc, bf16t* __restrict__ ob, int nb) {
  int total = na + nb;
  for (int i = blockIdx.x * blockDim.x + threadIdx.x; i < total; i += gridDim.x * blockDim.x) {
    if (i < na) oa[i] = f2bf(a[i]);
    else        ob[i - na] = f2bf(bsrc[i - na]);
  }
}

// ---------- QKV GEMM: D[o][n] = Wq[o,:]·xnT[n,:] + b[o] ----------
// q,k stored transposed [n][c]; v stored [c][n]
__global__ __launch_bounds__(256) void qkv_gemm_kernel(const bf16t* __restrict__ Wq,
                                                       const bf16t* __restrict__ XnT,
                                                       const float* __restrict__ qkvb,
                                                       bf16t* __restrict__ qT,
                                                       bf16t* __restrict__ kT,
                                                       bf16t* __restrict__ vC) {
  WAVE_COORDS();
  v8f acc[2][4]; zero_acc(acc);
  wave_tile_mma(Wq, kC, XnT, kC, kC, row0, col0, lane, acc);
  for (int si = 0; si < 2; ++si)
    for (int sj = 0; sj < 4; ++sj)
      for (int e = 0; e < 8; ++e) {
        int o = row0 + ACC_ROW(si, e);
        int n = col0 + ACC_COL(sj);
        float d = acc[si][sj][e] + qkvb[o];
        if (o < kC)          qT[(size_t)n * kC + o] = f2bf(d);
        else if (o < 2 * kC) kT[(size_t)n * kC + (o - kC)] = f2bf(d);
        else                 vC[(size_t)(o - 2 * kC) * kN + n] = f2bf(d);
      }
}

// ---------- Scores: S[i][j] = scale * qT[i,:]·kT[j,:] ----------
__global__ __launch_bounds__(256) void scores_kernel(const bf16t* __restrict__ qT,
                                                     const bf16t* __restrict__ kT,
                                                     float* __restrict__ S) {
  WAVE_COORDS();
  v8f acc[2][4]; zero_acc(acc);
  wave_tile_mma(qT, kC, kT, kC, kC, row0, col0, lane, acc);
  for (int si = 0; si < 2; ++si)
    for (int sj = 0; sj < 4; ++sj)
      for (int e = 0; e < 8; ++e) {
        int i = row0 + ACC_ROW(si, e);
        int j = col0 + ACC_COL(sj);
        S[(size_t)i * kN + j] = acc[si][sj][e] * kScale;
      }
}

// ---------- Row softmax: S f32 -> P bf16 ----------
__global__ __launch_bounds__(256) void softmax_kernel(const float* __restrict__ S,
                                                      bf16t* __restrict__ P) {
  const int i = blockIdx.x, tid = threadIdx.x;
  const float* row = S + (size_t)i * kN;
  __shared__ float sh[256];
  float m = -3.4e38f;
  for (int j = tid; j < kN; j += 256) m = fmaxf(m, row[j]);
  sh[tid] = m; __syncthreads();
  for (int off = 128; off > 0; off >>= 1) {
    if (tid < off) sh[tid] = fmaxf(sh[tid], sh[tid + off]);
    __syncthreads();
  }
  float rmax = sh[0];
  __syncthreads();
  float s = 0.f;
  for (int j = tid; j < kN; j += 256) s += __expf(row[j] - rmax);
  sh[tid] = s; __syncthreads();
  for (int off = 128; off > 0; off >>= 1) {
    if (tid < off) sh[tid] += sh[tid + off];
    __syncthreads();
  }
  float rinv = 1.0f / sh[0];
  bf16t* prow = P + (size_t)i * kN;
  for (int j = tid; j < kN; j += 256) prow[j] = f2bf(__expf(row[j] - rmax) * rinv);
}

// ---------- AV: oT[i][c] = P[i,:]·vC[c,:] ----------
__global__ __launch_bounds__(256) void av_kernel(const bf16t* __restrict__ P,
                                                 const bf16t* __restrict__ vC,
                                                 bf16t* __restrict__ oT) {
  WAVE_COORDS();
  v8f acc[2][4]; zero_acc(acc);
  wave_tile_mma(P, kN, vC, kN, kN, row0, col0, lane, acc);
  for (int si = 0; si < 2; ++si)
    for (int sj = 0; sj < 4; ++sj)
      for (int e = 0; e < 8; ++e) {
        int i = row0 + ACC_ROW(si, e);
        int c = col0 + ACC_COL(sj);
        oT[(size_t)i * kC + c] = f2bf(acc[si][sj][e]);
      }
}

// ---------- Proj + bias + residual: out[o][n] = Wp[o,:]·oT[n,:] + pb[o] + x[o][n] ----------
__global__ __launch_bounds__(256) void proj_kernel(const bf16t* __restrict__ Wp,
                                                   const bf16t* __restrict__ oT,
                                                   const float* __restrict__ pb,
                                                   const float* __restrict__ xb,
                                                   float* __restrict__ outb) {
  WAVE_COORDS();
  v8f acc[2][4]; zero_acc(acc);
  wave_tile_mma(Wp, kC, oT, kC, kC, row0, col0, lane, acc);
  for (int si = 0; si < 2; ++si)
    for (int sj = 0; sj < 4; ++sj)
      for (int e = 0; e < 8; ++e) {
        int o = row0 + ACC_ROW(si, e);
        int n = col0 + ACC_COL(sj);
        size_t idx = (size_t)o * kN + n;
        outb[idx] = acc[si][sj][e] + pb[o] + xb[idx];
      }
}

// ---------- host ----------
extern "C" void kernel_launch(void* const* d_in, const int* in_sizes, int n_in,
                              void* d_out, int out_size, void* d_ws, size_t ws_size,
                              hipStream_t stream) {
  (void)in_sizes; (void)n_in; (void)out_size; (void)ws_size;
  const float* x      = (const float*)d_in[0];
  const float* gn_w   = (const float*)d_in[1];
  const float* gn_b   = (const float*)d_in[2];
  const float* qkv_w  = (const float*)d_in[3];
  const float* qkv_b  = (const float*)d_in[4];
  const float* proj_w = (const float*)d_in[5];
  const float* proj_b = (const float*)d_in[6];
  float* out = (float*)d_out;

  char* p = (char*)d_ws;
  auto take = [&](size_t bytes) { char* r = p; p += (bytes + 255) & ~(size_t)255; return r; };
  bf16t* xnT = (bf16t*)take((size_t)kBatch * kN * kC * 2);  // normalized acts, [b][n][c]
  bf16t* wq  = (bf16t*)take((size_t)3 * kC * kC * 2);       // qkv_w bf16
  bf16t* wp  = (bf16t*)take((size_t)kC * kC * 2);           // proj_w bf16
  bf16t* qT  = (bf16t*)take((size_t)kN * kC * 2);           // per-batch Q^T [n][c]
  bf16t* kT  = (bf16t*)take((size_t)kN * kC * 2);           // per-batch K^T [n][c]
  bf16t* vC  = (bf16t*)take((size_t)kC * kN * 2);           // per-batch V   [c][n]
  float* S   = (float*)take((size_t)kN * kN * 4);           // per-batch scores
  bf16t* P   = (bf16t*)take((size_t)kN * kN * 2);           // per-batch softmax probs
  bf16t* oT  = (bf16t*)take((size_t)kN * kC * 2);           // per-batch attn out [n][c]

  wconv_kernel<<<1024, 256, 0, stream>>>(qkv_w, wq, 3 * kC * kC, proj_w, wp, kC * kC);
  gn_kernel<<<kBatch * kGroups, 256, 0, stream>>>(x, gn_w, gn_b, xnT);

  for (int b = 0; b < kBatch; ++b) {
    const bf16t* xb = xnT + (size_t)b * kN * kC;
    // block tile 128x128
    qkv_gemm_kernel<<<dim3(3 * kC / 128, kN / 128), 256, 0, stream>>>(wq, xb, qkv_b, qT, kT, vC);
    scores_kernel<<<dim3(kN / 128, kN / 128), 256, 0, stream>>>(qT, kT, S);
    softmax_kernel<<<kN, 256, 0, stream>>>(S, P);
    av_kernel<<<dim3(kN / 128, kC / 128), 256, 0, stream>>>(P, vC, oT);
    proj_kernel<<<dim3(kC / 128, kN / 128), 256, 0, stream>>>(
        wp, oT, proj_b, x + (size_t)b * kC * kN, out + (size_t)b * kC * kN);
  }
}
